// ResNet_58205396795418
// MI455X (gfx1250) — matrix operand order repaired and checked by reading, not compiled, not run on hardware
//
#include <hip/hip_runtime.h>
#include <hip/hip_bf16.h>
#include <cstdint>
#include <cstddef>

// ---------------------------------------------------------------------------
// MI455X (gfx1250) sparse-ResNet forward.
//  - Activations: f16, NDHWC so WMMA A-fragments load as two contiguous b128s
//    per lane per 32-wide K chunk (K tap-major / channel-minor).
//  - 3^3 convs: implicit GEMM on v_wmma_f32_16x16x32_f16. One wave owns FOUR
//    16-voxel M tiles x one 16-cout N tile (4 accumulators). Packed B
//    fragments are staged in LDS once per block via the Tensor Data Mover
//    (tensor_load_to_lds + s_wait_tensorcnt), then read with ds_load_b128
//    -> 4 WMMAs per B fetch.
//  - Masked BN: one-pass sum/sumsq with f32 global atomics; var=E[x2]-E[x]^2.
//  - All power-of-two dims passed as log2 -> no integer division in kernels.
//  - Workspace: masks ~9.6MB @0, stats @~9.59MB, packed weights @10MB,
//    act buf A @16MB (64MB), buf B @80MB (64MB)  => ~144MB total.
// ---------------------------------------------------------------------------

typedef _Float16 h8   __attribute__((ext_vector_type(8)));
typedef _Float16 v16h __attribute__((ext_vector_type(16)));
typedef float    v8f  __attribute__((ext_vector_type(8)));
typedef unsigned int u32x4 __attribute__((ext_vector_type(4)));
typedef int          i32x8 __attribute__((ext_vector_type(8)));
typedef int          i32x4 __attribute__((ext_vector_type(4)));

__global__ void k_zero(float* __restrict__ p, int n) {
  int i = blockIdx.x * blockDim.x + threadIdx.x;
  if (i < n) p[i] = 0.f;
}

__global__ void k_mask_to_float(const int* __restrict__ m, float* __restrict__ o, int n) {
  int i = blockIdx.x * blockDim.x + threadIdx.x;
  if (i < n) o[i] = (float)m[i];
}

__global__ void k_sum(const float* __restrict__ in, int n, float* __restrict__ out) {
  __shared__ float sh[256];
  float s = 0.f;
  for (int i = blockIdx.x * blockDim.x + threadIdx.x; i < n; i += blockDim.x * gridDim.x)
    s += in[i];
  sh[threadIdx.x] = s;
  __syncthreads();
  for (int o = 128; o > 0; o >>= 1) {
    if ((int)threadIdx.x < o) sh[threadIdx.x] += sh[threadIdx.x + o];
    __syncthreads();
  }
  if (threadIdx.x == 0) atomicAdd(out, sh[0]);
}

__global__ void k_pool_m(const float* __restrict__ in, float* __restrict__ out, int lSo) {
  int i = blockIdx.x * blockDim.x + threadIdx.x;
  int So = 1 << lSo;
  if (i >= (1 << (3 * lSo))) return;
  int x = i & (So - 1), y = (i >> lSo) & (So - 1), z = i >> (2 * lSo);
  int Si = So << 1;
  float mx = 0.f;
  for (int dz = 0; dz < 2; dz++)
    for (int dy = 0; dy < 2; dy++)
      for (int dx = 0; dx < 2; dx++)
        mx = fmaxf(mx, in[((z * 2 + dz) * Si + (y * 2 + dy)) * Si + (x * 2 + dx)]);
  out[i] = mx;
}

// Pack OIDHW f32 weights into WMMA B-fragment order:
// element i -> (nt, k-chunk, lane, e); value = W[cout = nt*16 + lane%16]
//   [K = k*32 + (lane/16)*16 + e], K decomposed tap-major / cin-minor.
__global__ void k_pack_w(const float* __restrict__ w, _Float16* __restrict__ o,
                         int Cout, int Cin, int NC, long total) {
  long i = (long)blockIdx.x * blockDim.x + threadIdx.x;
  if (i >= total) return;
  int  e  = (int)(i & 15);
  int  l  = (int)((i >> 4) & 31);
  long r  = i >> 9;
  int  k  = (int)(r % NC);
  int  nt = (int)(r / NC);
  int  n  = (nt << 4) + (l & 15);
  int  K  = (k << 5) + ((l >> 4) << 4) + e;
  int  Ktot = Cin * 27;
  float v = 0.f;
  if (n < Cout && K < Ktot) {
    int tap = K / Cin;
    int ci  = K - tap * Cin;
    v = w[((long)n * Cin + ci) * 27 + tap];
  }
  o[i] = (_Float16)v;
}

// Initial 5^3 conv, 1 -> 8 channels, masked (submanifold). VALU-bound, tiny
// vs the WMMA levels. Output f16 NDHWC at 128^3 x 8.
__global__ void __launch_bounds__(256) k_conv_init(
    const float* __restrict__ x, const float* __restrict__ msk,
    const float* __restrict__ w, _Float16* __restrict__ out) {
  int v = blockIdx.x * blockDim.x + threadIdx.x;
  if (v >= 128 * 128 * 128) return;
  int X = v & 127, Y = (v >> 7) & 127, Z = v >> 14;
  float mv = msk[v];
  float acc[8] = {0.f, 0.f, 0.f, 0.f, 0.f, 0.f, 0.f, 0.f};
  if (mv != 0.f) {
    for (int kd = 0; kd < 5; kd++) {
      int Zz = Z + kd - 2; if ((unsigned)Zz >= 128u) continue;
      for (int kh = 0; kh < 5; kh++) {
        int Yy = Y + kh - 2; if ((unsigned)Yy >= 128u) continue;
        for (int kw = 0; kw < 5; kw++) {
          int Xx = X + kw - 2; if ((unsigned)Xx >= 128u) continue;
          float xv = x[(Zz << 14) + (Yy << 7) + Xx];
          int tap = (kd * 5 + kh) * 5 + kw;
          #pragma unroll
          for (int c = 0; c < 8; c++) acc[c] += w[c * 125 + tap] * xv;
        }
      }
    }
  }
  #pragma unroll
  for (int c = 0; c < 8; c++) out[((size_t)v << 3) + c] = (_Float16)acc[c];
}

// Implicit-GEMM 3^3 conv on v_wmma_f32_16x16x32_f16.
// Wave = 4 voxel tiles (64 voxels along linear index) x 16 cout.
// Block (8 waves) is N-tile uniform; packed B fragments for this N tile are
// DMA'd into LDS by the Tensor Data Mover (wave 0 issues; TDM ignores EXEC),
// then each chunk's B is read via ds_load_b128 and multiplied against 4 A
// fragments (4 WMMAs / B fetch).
// Grids divide exactly (no early exit): EXEC all-ones at every WMMA.
__global__ void __launch_bounds__(256) k_conv_wmma(
    const _Float16* __restrict__ act, const _Float16* __restrict__ wp,
    const float* __restrict__ msk, _Float16* __restrict__ outp,
    int lS, int lCi, int Cout, int lCo, int NC, int lMT) {
  extern __shared__ __align__(16) _Float16 lw[];   // NC*512 f16
  const int S = 1 << lS;
  int wave = blockIdx.x * (blockDim.x >> 5) + (threadIdx.x >> 5);
  int lane = threadIdx.x & 31;
  int nt   = wave >> lMT;              // (1<<lMT) multiple of 8 => block-uniform
  int mt   = wave & ((1 << lMT) - 1);
  int v00  = mt << 6;                  // 64 voxels per wave
  int half = lane >> 4;
  int mrow = lane & 15;
  const int Cin  = 1 << lCi;
  const int Ktot = 27 << lCi;

  // ---- TDM: stage this N-tile's packed weights into LDS ----
  if ((threadIdx.x >> 5) == 0) {       // one wave issues the descriptor
    int ntb = (int)((blockIdx.x * (blockDim.x >> 5)) >> lMT);  // block-uniform
    unsigned long long ga =
        (unsigned long long)(const void*)(wp + (size_t)ntb * NC * 512);
    int n8 = NC << 7;                  // tile length in 8-byte units
    // D# group 0: count=1 | lds_addr=0 | global_addr[56:0] | type=2
    u32x4 g0;
    g0[0] = 1u;
    g0[1] = 0u;
    g0[2] = (unsigned)(ga & 0xffffffffu);
    g0[3] = (unsigned)((ga >> 32) & 0x01ffffffu) | (2u << 30);
    // D# group 1: data_size=8B, 1-D tensor dim0=n8, tile_dim0=n8,
    //             tile_dim1=1, dim0_stride=n8, no multicast / barrier / pad.
    i32x8 g1 = {};
    g1[0] = (3 << 16);                 // data_size code 3 = 8 bytes
    g1[1] = (n8 & 0xffff) << 16;       // tensor_dim0[15:0]
    g1[2] = ((n8 >> 16) & 0xffff) | (1 << 16);  // tensor_dim0[31:16], dim1=1
    g1[3] = (n8 & 0xffff) << 16;       // tile_dim0
    g1[4] = 1;                         // tile_dim1 = 1
    g1[5] = n8;                        // tensor_dim0_stride[31:0]
    i32x4 gz4 = {};
    i32x8 gz8 = {};
    // amdgpu-toolchain (clang-23) 6-arg form: (g0, g1, g2, g3, g4, cpol)
    __builtin_amdgcn_tensor_load_to_lds(g0, g1, gz4, gz4, gz8, 0);
    __builtin_amdgcn_s_wait_tensorcnt((short)0);
  }
  __syncthreads();

  // per-tile lane-row coordinates
  int X0[4], Y0[4], Z0[4];
  #pragma unroll
  for (int t = 0; t < 4; t++) {
    int vm = v00 + (t << 4) + mrow;
    X0[t] = vm & (S - 1);
    Y0[t] = (vm >> lS) & (S - 1);
    Z0[t] = vm >> (2 * lS);
  }

  v8f acc[4] = {{}, {}, {}, {}};
  for (int k = 0; k < NC; ++k) {
    union { v16h v; h8 r[2]; } B;
    const h8* bp = (const h8*)(lw + ((size_t)k << 9) + (lane << 4));
    B.r[0] = bp[0];
    B.r[1] = bp[1];
    #pragma unroll
    for (int t = 0; t < 4; t++) {
      union { v16h v; h8 r[2]; } A;
      #pragma unroll
      for (int rr = 0; rr < 2; ++rr) {
        h8 val = {};
        int K0 = (k << 5) + (half << 4) + (rr << 3);
        if (K0 < Ktot) {
          int tap = K0 >> lCi;
          int ci  = K0 & (Cin - 1);
          int kw = tap % 3, kh = (tap / 3) % 3, kd = tap / 9;   // consts: mul/shift
          int X = X0[t] + kw - 1, Y = Y0[t] + kh - 1, Z = Z0[t] + kd - 1;
          if ((unsigned)X < (unsigned)S && (unsigned)Y < (unsigned)S &&
              (unsigned)Z < (unsigned)S) {
            size_t off = ((((size_t)Z << lS) + (size_t)Y) << lS) + (size_t)X;
            val = *(const h8*)(act + (off << lCi) + ci);   // global_load_b128
          }
        }
        A.r[rr] = val;
      }
      acc[t] = __builtin_amdgcn_wmma_f32_16x16x32_f16(
          false, A.v, false, B.v, (short)0, acc[t], false, false);
    }
  }

  int n = (nt << 4) + mrow;
  if (n < Cout) {
    #pragma unroll
    for (int t = 0; t < 4; t++) {
      #pragma unroll
      for (int r = 0; r < 8; ++r) {
        int vv = v00 + (t << 4) + (half << 3) + r;   // D layout: M = r + 8*half
        float mv = msk[vv];
        outp[((size_t)vv << lCo) + n] = (_Float16)(acc[t][r] * mv);
      }
    }
  }
}

// Per-channel sum & sumsq over masked pre-BN activations (already x*m).
__global__ void k_stats(const _Float16* __restrict__ p, int V, int Cout, int lCo,
                        float* __restrict__ sums) {
  int c = threadIdx.x & (Cout - 1);
  int g = threadIdx.x >> lCo;
  int rpb = blockDim.x >> lCo;
  float s = 0.f, ss = 0.f;
  for (int v = blockIdx.x * rpb + g; v < V; v += gridDim.x * rpb) {
    float xv = (float)p[((size_t)v << lCo) + c];
    s += xv; ss += xv * xv;
  }
  atomicAdd(&sums[c], s);
  atomicAdd(&sums[128 + c], ss);
}

__global__ void k_bn_relu(const _Float16* __restrict__ p, const float* __restrict__ msk,
                          const float* __restrict__ sums, const float* __restrict__ cntp,
                          const float* __restrict__ gam, const float* __restrict__ bet,
                          _Float16* __restrict__ out, int total, int Cout, int lCo) {
  int i = blockIdx.x * blockDim.x + threadIdx.x;
  if (i >= total) return;
  int c = i & (Cout - 1);
  int v = i >> lCo;
  float n = fmaxf(cntp[0], 1.f);
  float mean = sums[c] / n;
  float var  = fmaxf(sums[128 + c] / n - mean * mean, 0.f);
  float sc = gam[c] * rsqrtf(var + 1e-4f);
  float y = (((float)p[i]) - mean) * sc + bet[c];
  y = fmaxf(y * msk[v], 0.f);
  out[i] = (_Float16)y;
}

__global__ void k_pool_x(const _Float16* __restrict__ in, const float* __restrict__ mo,
                         _Float16* __restrict__ out, int lSo, int lC, int total) {
  int i = blockIdx.x * blockDim.x + threadIdx.x;
  if (i >= total) return;
  int So = 1 << lSo;
  int c = i & ((1 << lC) - 1);
  int v = i >> lC;
  int x = v & (So - 1), y = (v >> lSo) & (So - 1), z = v >> (2 * lSo);
  int Si = So << 1;
  float mx = 0.f;                          // inputs are post-ReLU (>= 0)
  for (int dz = 0; dz < 2; dz++)
    for (int dy = 0; dy < 2; dy++)
      for (int dx = 0; dx < 2; dx++) {
        size_t idx = ((size_t)((z * 2 + dz) * Si + (y * 2 + dy)) * Si + (x * 2 + dx));
        mx = fmaxf(mx, (float)in[(idx << lC) + c]);
      }
  out[i] = (_Float16)(mx * mo[v]);
}

// pad+1x1 conv+mean collapses to softmax(W . (sum over voxels x) / 1000).
__global__ void k_final(const _Float16* __restrict__ act, const float* __restrict__ wb,
                        float* __restrict__ out) {
  __shared__ float s[128];
  __shared__ float logit[10];
  int t = threadIdx.x;                    // blockDim == 128
  float a = 0.f;
  for (int v = 0; v < 512; ++v) a += (float)act[v * 128 + t];
  s[t] = a;
  __syncthreads();
  if (t < 10) {
    float l = 0.f;
    for (int c = 0; c < 128; c++) l += wb[t * 128 + c] * s[c];
    logit[t] = l * 0.001f;                // mean over padded 10^3 = /1000
  }
  __syncthreads();
  if (t == 0) {
    float mx = logit[0];
    for (int j = 1; j < 10; j++) mx = fmaxf(mx, logit[j]);
    float e[10]; float se = 0.f;
    for (int j = 0; j < 10; j++) { e[j] = expf(logit[j] - mx); se += e[j]; }
    for (int j = 0; j < 10; j++) out[j] = e[j] / se;
  }
}

extern "C" void kernel_launch(void* const* d_in, const int* in_sizes, int n_in,
                              void* d_out, int out_size, void* d_ws, size_t ws_size,
                              hipStream_t stream) {
  (void)in_sizes; (void)n_in; (void)out_size; (void)ws_size;

  // --- input pointers (setup_inputs() insertion order, params flattened) ---
  const float* x      = (const float*)d_in[0];
  const int*   mint   = (const int*)  d_in[1];
  const float* w_init = (const float*)d_in[2];
  const float *W[4][4], *G[4][4], *Bt[4][4];
  int idx = 3;
  for (int L = 0; L < 4; L++) {
    for (int j = 0; j < 4; j++) W[L][j]  = (const float*)d_in[idx++];
    for (int j = 0; j < 4; j++) G[L][j]  = (const float*)d_in[idx++];
    for (int j = 0; j < 4; j++) Bt[L][j] = (const float*)d_in[idx++];
  }
  const float* wb = (const float*)d_in[idx];

  // --- workspace carving ---
  char* ws = (char*)d_ws;
  float* mbuf[5];
  mbuf[0] = (float*)(ws + 0);                       // 128^3 f32 = 8 MB
  mbuf[1] = (float*)(ws + (size_t)(8u << 20));      // 64^3
  mbuf[2] = (float*)(ws + 9437184u);                // 32^3
  mbuf[3] = (float*)(ws + 9568256u);                // 16^3
  mbuf[4] = (float*)(ws + 9584640u);                // 8^3
  float* cnt   = (float*)(ws + 9586688u);           // 4 f32 (per-level counts)
  float* stats = (float*)(ws + 9587200u);           // 256 f32 (sum | sumsq)
  char*  wpackBase = ws + (size_t)(10u << 20);      // packed f16 weights (~1.5MB)
  _Float16* bufA = (_Float16*)(ws + (size_t)(16u << 20));  // 64 MB
  _Float16* bufB = (_Float16*)(ws + (size_t)(80u << 20));  // 64 MB

  const int V0 = 128 * 128 * 128;

  // 1) mask -> float, pooled mask pyramid, active-site counts per level
  k_mask_to_float<<<(V0 + 255) / 256, 256, 0, stream>>>(mint, mbuf[0], V0);
  for (int L = 1; L <= 4; L++) {
    int lSo = 7 - L;
    int n   = 1 << (3 * lSo);
    k_pool_m<<<(n + 255) / 256, 256, 0, stream>>>(mbuf[L - 1], mbuf[L], lSo);
  }
  k_zero<<<1, 256, 0, stream>>>(cnt, 4);
  for (int L = 0; L < 4; L++) {
    int n = 1 << (3 * (7 - L));
    int grid = (n + 255) / 256; if (grid > 256) grid = 256;
    k_sum<<<grid, 256, 0, stream>>>(mbuf[L], n, cnt + L);
  }

  // 2) pack all conv weights to f16 B-fragment order
  _Float16* wpk[4][4];
  size_t woff = 0;
  for (int L = 0; L < 4; L++) {
    int c0 = 8 << L;
    for (int j = 0; j < 4; j++) {
      int Cin  = c0;
      int Cout = (j == 3) ? 2 * c0 : c0;
      int NC = (Cin * 27 + 31) / 32;
      int NT = (Cout + 15) / 16;
      wpk[L][j] = (_Float16*)(wpackBase + woff);
      long total = (long)NT * NC * 512;
      woff += (size_t)total * 2;
      woff = (woff + 255) & ~(size_t)255;
      k_pack_w<<<(int)((total + 255) / 256), 256, 0, stream>>>(
          W[L][j], wpk[L][j], Cout, Cin, NC, total);
    }
  }

  // 3) initial 5^3 submanifold conv -> f16 NDHWC in bufA
  k_conv_init<<<(V0 + 255) / 256, 256, 0, stream>>>(x, mbuf[0], w_init, bufA);

  // 4) levels: (wmma conv -> stats -> bn/relu) x4, then pool
  _Float16* cur = bufA;
  _Float16* tmp = bufB;
  for (int L = 0; L < 4; L++) {
    int lS = 7 - L;
    int V  = 1 << (3 * lS);
    int c0 = 8 << L;
    for (int j = 0; j < 4; j++) {
      int Cin  = c0;
      int Cout = (j == 3) ? 2 * c0 : c0;
      int lCi  = 3 + L;
      int lCo  = (j == 3) ? 4 + L : 3 + L;
      int NC = (Cin * 27 + 31) / 32;
      int NT = (Cout + 15) / 16;
      int lMT = 3 * lS - 6;            // log2(M-groups per N tile), >= 6

      k_zero<<<1, 256, 0, stream>>>(stats, 256);

      int totalWaves = (V >> 6) * NT;  // 4 voxel tiles per wave
      int blocks = totalWaves >> 3;    // 8 waves per block, divides exactly
      size_t shbytes = (size_t)NC * 1024;
      k_conv_wmma<<<blocks, 256, shbytes, stream>>>(
          cur, wpk[L][j], mbuf[L], tmp, lS, lCi, Cout, lCo, NC, lMT);

      int rpb = 256 >> lCo;
      int sgrid = (V + rpb - 1) / rpb; if (sgrid > 1024) sgrid = 1024;
      k_stats<<<sgrid, 256, 0, stream>>>(tmp, V, Cout, lCo, stats);

      int tot = V << lCo;
      k_bn_relu<<<(tot + 255) / 256, 256, 0, stream>>>(
          tmp, mbuf[L], stats, cnt + L, G[L][j], Bt[L][j], cur, tot, Cout, lCo);
    }
    // maxpool x (2^3) and apply pooled mask; swap buffers
    int lSo = lS - 1;
    int lC  = 4 + L;                   // channels after block4 = 2*c0
    int tot = (1 << (3 * lSo)) << lC;
    k_pool_x<<<(tot + 255) / 256, 256, 0, stream>>>(cur, mbuf[L + 1], tmp, lSo, lC, tot);
    _Float16* t2 = cur; cur = tmp; tmp = t2;
  }

  // 5) bottleneck + global mean + softmax
  k_final<<<1, 128, 0, stream>>>(cur, wb, (float*)d_out);
}
